// WaterLevelGCN_25288767438901
// MI455X (gfx1250) — matrix operand order repaired and checked by reading, not compiled, not run on hardware
//
#include <hip/hip_runtime.h>
#include <hip/hip_bf16.h>

typedef __attribute__((ext_vector_type(16))) _Float16 v16h;
typedef __attribute__((ext_vector_type(8)))  _Float16 v8h;
typedef __attribute__((ext_vector_type(8)))  float    v8f;

#define NODES  4096
#define DMODEL 128
#define NHEAD  4
#define DHEAD  32
#define DFFN   2048
#define HIDDEN 256
#define OUTD   64
#define NEDGE  131072

static __device__ inline v16h wl_make_afrag(v8h lo, v8h hi) {
  v16h a;
#pragma unroll
  for (int i = 0; i < 8; ++i) { a[i] = lo[i]; a[i + 8] = hi[i]; }
  return a;
}

#define WL_ZERO8 {0.f,0.f,0.f,0.f,0.f,0.f,0.f,0.f}

// ---------------------------------------------------------------------------
// WMMA GEMM: C[4096, Nout] = A[4096,K](f16) * W[Nout,K]^T(f16) (+bias)(+relu)
// Block = 8 waves = 256 rows x 32 cols. Wave tile = 32x32 (4 accumulators).
// Weight tile (32 x KC) is staged into LDS once per block via
// GLOBAL_LOAD_ASYNC_TO_LDS_B128 (ASYNCcnt) and shared by all 8 waves.
// Fragment k order (A and B): contiguous, k = 16*(lane/16) + s (wave32).
// ---------------------------------------------------------------------------
__global__ __launch_bounds__(256) void wl_gemm_kernel(
    const _Float16* __restrict__ A, const _Float16* __restrict__ W,
    const float* __restrict__ bias, float* __restrict__ Cf,
    _Float16* __restrict__ Ch, int Nout, int K, int relu)
{
  __shared__ __align__(32) _Float16 wtile[32 * 256];   // 16 KB
  const int tid  = threadIdx.x;
  const int lane = tid & 31;
  const int wave = tid >> 5;
  const int row0 = blockIdx.y * 256 + wave * 32;
  const int col0 = blockIdx.x * 32;
  const int m    = lane & 15;
  const int hsel = lane >> 4;       // 0: lanes 0-15, 1: lanes 16-31
  const int kbA  = hsel * 16;       // A-frag K base
  const int kbB  = hsel * 16;       // B-frag K base
  const int KC   = (K < 256) ? K : 256;   // K chunk staged in LDS
  const int kcv  = KC >> 3;               // b128 vectors per weight row

  v8f acc00 = WL_ZERO8, acc01 = WL_ZERO8, acc10 = WL_ZERO8, acc11 = WL_ZERO8;
  const _Float16* Arow0 = A + (size_t)(row0 + m) * K;
  const _Float16* Arow1 = A + (size_t)(row0 + 16 + m) * K;
  const unsigned lbase = (unsigned)(uintptr_t)&wtile[0];  // LDS byte offset

  for (int kc0 = 0; kc0 < K; kc0 += KC) {
    __syncthreads();   // previous chunk fully consumed by all waves
    // ---- async stage W[col0..col0+31][kc0..kc0+KC) -> LDS ----
    for (int i = tid; i < 32 * kcv; i += 256) {
      const int r = i / kcv;
      const int c = i - r * kcv;
      const _Float16* gp = W + (size_t)(col0 + r) * K + kc0 + c * 8;
      const unsigned la = lbase + (unsigned)(i * 16);
      asm volatile("global_load_async_to_lds_b128 %0, %1, off"
                   :: "v"(la), "v"((unsigned long long)(uintptr_t)gp)
                   : "memory");
    }
    asm volatile("s_wait_asynccnt 0" ::: "memory");
    __syncthreads();   // chunk visible to all waves
    // ---- consume chunk ----
    for (int k0 = 0; k0 < KC; k0 += 32) {
      v16h a0 = wl_make_afrag(*(const v8h*)(Arow0 + kc0 + k0 + kbA),
                              *(const v8h*)(Arow0 + kc0 + k0 + 8 + kbA));
      v16h a1 = wl_make_afrag(*(const v8h*)(Arow1 + kc0 + k0 + kbA),
                              *(const v8h*)(Arow1 + kc0 + k0 + 8 + kbA));
      v16h b0 = *(const v16h*)&wtile[m * KC + k0 + kbB];
      v16h b1 = *(const v16h*)&wtile[(16 + m) * KC + k0 + kbB];
      acc00 = __builtin_amdgcn_wmma_f32_16x16x32_f16(false, a0, false, b0, (short)0, acc00, false, false);
      acc01 = __builtin_amdgcn_wmma_f32_16x16x32_f16(false, a0, false, b1, (short)0, acc01, false, false);
      acc10 = __builtin_amdgcn_wmma_f32_16x16x32_f16(false, a1, false, b0, (short)0, acc10, false, false);
      acc11 = __builtin_amdgcn_wmma_f32_16x16x32_f16(false, a1, false, b1, (short)0, acc11, false, false);
    }
  }

  const int n0 = col0 + m;
  const float bv0 = bias ? bias[n0] : 0.0f;
  const float bv1 = bias ? bias[n0 + 16] : 0.0f;
  const int rb0 = row0 + hsel * 8;
  const int rb1 = row0 + 16 + hsel * 8;
#pragma unroll
  for (int r = 0; r < 8; ++r) {
    float c00 = acc00[r] + bv0, c01 = acc01[r] + bv1;
    float c10 = acc10[r] + bv0, c11 = acc11[r] + bv1;
    if (relu) {
      c00 = fmaxf(c00, 0.f); c01 = fmaxf(c01, 0.f);
      c10 = fmaxf(c10, 0.f); c11 = fmaxf(c11, 0.f);
    }
    const size_t i00 = (size_t)(rb0 + r) * Nout + n0;
    const size_t i10 = (size_t)(rb1 + r) * Nout + n0;
    if (Cf) { Cf[i00] = c00; Cf[i00 + 16] = c01; Cf[i10] = c10; Cf[i10 + 16] = c11; }
    if (Ch) { Ch[i00] = (_Float16)c00; Ch[i00 + 16] = (_Float16)c01;
              Ch[i10] = (_Float16)c10; Ch[i10 + 16] = (_Float16)c11; }
  }
}

// ---------------------------------------------------------------------------
// Flash-style attention. qkv[N,384] f16 (q|k|v), vt[128][N] f16 transposed V.
// One wave = 16 query rows of one head. Online softmax in fp32; P transposed
// to A-fragment layout through per-wave LDS.
// ---------------------------------------------------------------------------
__global__ __launch_bounds__(256) void wl_attn_kernel(
    const _Float16* __restrict__ qkv, const _Float16* __restrict__ vt,
    _Float16* __restrict__ ctx)
{
  __shared__ __align__(32) _Float16 pbuf[8][16 * 32];
  const int lane = threadIdx.x & 31;
  const int wave = threadIdx.x >> 5;
  const int gw   = blockIdx.x * 8 + wave;   // 0..1023
  const int head = gw >> 8;                 // 0..3
  const int rt   = gw & 255;                // row tile
  const int row0 = rt * 16;
  const int m    = lane & 15;
  const int hsel = lane >> 4;
  const int kbA  = hsel * 16;
  const int kbB  = hsel * 16;
  const int LDQ  = 3 * DMODEL;              // 384

  const _Float16* qrow = qkv + (size_t)(row0 + m) * LDQ + head * DHEAD;
  v16h qa = wl_make_afrag(*(const v8h*)(qrow + kbA),
                          *(const v8h*)(qrow + 8 + kbA));

  float mrun[8], lrun[8];
#pragma unroll
  for (int r = 0; r < 8; ++r) { mrun[r] = -1e30f; lrun[r] = 0.0f; }
  v8f o0 = WL_ZERO8, o1 = WL_ZERO8;

  const _Float16* kmat = qkv + DMODEL + head * DHEAD;
  const _Float16* vlo  = vt + (size_t)(head * DHEAD + m)      * NODES;
  const _Float16* vhi  = vt + (size_t)(head * DHEAD + 16 + m) * NODES;
  _Float16* pb = &pbuf[wave][0];
  const float SCALE = 0.17677669529663687f;  // 1/sqrt(32)

  for (int j0 = 0; j0 < NODES; j0 += 32) {
    v16h b0 = *(const v16h*)(kmat + (size_t)(j0 + m) * LDQ + kbB);
    v16h b1 = *(const v16h*)(kmat + (size_t)(j0 + 16 + m) * LDQ + kbB);
    v8f s0 = WL_ZERO8, s1 = WL_ZERO8;
    s0 = __builtin_amdgcn_wmma_f32_16x16x32_f16(false, qa, false, b0, (short)0, s0, false, false);
    s1 = __builtin_amdgcn_wmma_f32_16x16x32_f16(false, qa, false, b1, (short)0, s1, false, false);

#pragma unroll
    for (int r = 0; r < 8; ++r) {
      float a0 = s0[r] * SCALE;
      float a1 = s1[r] * SCALE;
      float t = fmaxf(a0, a1);
      t = fmaxf(t, __shfl_xor(t, 1, 32));
      t = fmaxf(t, __shfl_xor(t, 2, 32));
      t = fmaxf(t, __shfl_xor(t, 4, 32));
      t = fmaxf(t, __shfl_xor(t, 8, 32));      // row max within 16-lane half
      float mn = fmaxf(mrun[r], t);
      float alpha = __expf(mrun[r] - mn);
      mrun[r] = mn;
      float p0 = __expf(a0 - mn);
      float p1 = __expf(a1 - mn);
      float rs = p0 + p1;
      rs += __shfl_xor(rs, 1, 32);
      rs += __shfl_xor(rs, 2, 32);
      rs += __shfl_xor(rs, 4, 32);
      rs += __shfl_xor(rs, 8, 32);
      lrun[r] = lrun[r] * alpha + rs;
      o0[r] *= alpha;
      o1[r] *= alpha;
      const int row = r + hsel * 8;
      pb[row * 32 + m]      = (_Float16)p0;
      pb[row * 32 + 16 + m] = (_Float16)p1;
    }
    // same-wave LDS RAW: DS ops are in-order; wait + compiler fence
    asm volatile("s_wait_dscnt 0" ::: "memory");

    const _Float16* prow = pb + m * 32;
    v16h pa = wl_make_afrag(*(const v8h*)(prow + kbA),
                            *(const v8h*)(prow + 8 + kbA));
    v16h vb0 = *(const v16h*)(vlo + j0 + kbB);
    v16h vb1 = *(const v16h*)(vhi + j0 + kbB);
    o0 = __builtin_amdgcn_wmma_f32_16x16x32_f16(false, pa, false, vb0, (short)0, o0, false, false);
    o1 = __builtin_amdgcn_wmma_f32_16x16x32_f16(false, pa, false, vb1, (short)0, o1, false, false);
    asm volatile("" ::: "memory");
  }

  const int n0 = head * DHEAD;
#pragma unroll
  for (int r = 0; r < 8; ++r) {
    const int row = row0 + hsel * 8 + r;
    const float inv = 1.0f / lrun[r];
    ctx[(size_t)row * DMODEL + n0 + m]      = (_Float16)(o0[r] * inv);
    ctx[(size_t)row * DMODEL + n0 + 16 + m] = (_Float16)(o1[r] * inv);
  }
}

// ---------------------------------------------------------------------------
// LayerNorm(a + b) * g + beta; one wave per 128-wide row. f32 + f16 outputs.
// ---------------------------------------------------------------------------
__global__ __launch_bounds__(256) void wl_ln_kernel(
    const float* __restrict__ a, const float* __restrict__ b,
    const float* __restrict__ g, const float* __restrict__ beta,
    float* __restrict__ of, _Float16* __restrict__ oh)
{
  const int lane = threadIdx.x & 31;
  const int wave = threadIdx.x >> 5;
  const int row  = blockIdx.x * 8 + wave;
  const float* pa = a + (size_t)row * DMODEL;
  const float* pb = b + (size_t)row * DMODEL;
  float v[4];
  float s = 0.f, sq = 0.f;
#pragma unroll
  for (int i = 0; i < 4; ++i) {
    v[i] = pa[lane * 4 + i] + pb[lane * 4 + i];
    s += v[i]; sq += v[i] * v[i];
  }
#pragma unroll
  for (int off = 1; off < 32; off <<= 1) {
    s  += __shfl_xor(s,  off, 32);
    sq += __shfl_xor(sq, off, 32);
  }
  const float mu  = s * (1.0f / 128.0f);
  const float var = sq * (1.0f / 128.0f) - mu * mu;
  const float rs  = rsqrtf(var + 1e-5f);
#pragma unroll
  for (int i = 0; i < 4; ++i) {
    const int c = lane * 4 + i;
    const float o = (v[i] - mu) * rs * g[c] + beta[c];
    if (of) of[(size_t)row * DMODEL + c] = o;
    if (oh) oh[(size_t)row * DMODEL + c] = (_Float16)o;
  }
}

// --------------------------- small utility kernels -------------------------
__global__ void wl_cvt_f32_f16(const float* __restrict__ in,
                               _Float16* __restrict__ out, int n) {
  int i = blockIdx.x * 256 + threadIdx.x;
  if (i < n) out[i] = (_Float16)in[i];
}

__global__ void wl_vtrans(const _Float16* __restrict__ qkv,
                          _Float16* __restrict__ vt) {
  int i = blockIdx.x * 256 + threadIdx.x;       // over NODES*128
  if (i >= NODES * DMODEL) return;
  int n = i >> 7, c = i & 127;
  vt[(size_t)c * NODES + n] = qkv[(size_t)n * (3 * DMODEL) + 2 * DMODEL + c];
}

__global__ void wl_fill(float* __restrict__ p, float v, int n) {
  int i = blockIdx.x * 256 + threadIdx.x;
  if (i < n) p[i] = v;
}

__global__ void wl_deg_kernel(const int* __restrict__ dst,
                              float* __restrict__ deg) {
  int e = blockIdx.x * 256 + threadIdx.x;
  if (e < NEDGE)
    __hip_atomic_fetch_add(&deg[dst[e]], 1.0f, __ATOMIC_RELAXED,
                           __HIP_MEMORY_SCOPE_AGENT);
}

__global__ void wl_dinv_kernel(const float* __restrict__ deg,
                               float* __restrict__ dinv) {
  int i = blockIdx.x * 256 + threadIdx.x;
  if (i < NODES) dinv[i] = rsqrtf(deg[i]);
}

// out[n][c] = hpre[n][c]*dinv[n]^2 + bias[c]   (self loop + bias)
__global__ void wl_gcn_init(const float* __restrict__ hpre,
                            const float* __restrict__ dinv,
                            const float* __restrict__ bias,
                            float* __restrict__ hsum, int dim) {
  int i = blockIdx.x * 256 + threadIdx.x;
  if (i >= NODES * dim) return;
  int n = i / dim, c = i - n * dim;
  float di = dinv[n];
  hsum[i] = hpre[i] * di * di + bias[c];
}

// out[dst] += hpre[src] * dinv[src]*dinv[dst], 4 features per thread
__global__ void wl_gcn_scatter(const float* __restrict__ hpre,
                               const float* __restrict__ dinv,
                               const int* __restrict__ src,
                               const int* __restrict__ dst,
                               float* __restrict__ hsum, int dim) {
  int t = blockIdx.x * 256 + threadIdx.x;
  int per = dim >> 2;
  if (t >= NEDGE * per) return;
  int e  = t / per;
  int c4 = (t - e * per) << 2;
  int s = src[e], d = dst[e];
  float w = dinv[s] * dinv[d];
  const float* hs = hpre + (size_t)s * dim + c4;
  float* hd = hsum + (size_t)d * dim + c4;
#pragma unroll
  for (int i = 0; i < 4; ++i)
    __hip_atomic_fetch_add(hd + i, hs[i] * w, __ATOMIC_RELAXED,
                           __HIP_MEMORY_SCOPE_AGENT);
}

__global__ void wl_relu_cvt(const float* __restrict__ in,
                            _Float16* __restrict__ out, int n) {
  int i = blockIdx.x * 256 + threadIdx.x;
  if (i < n) out[i] = (_Float16)fmaxf(in[i], 0.0f);
}

// ---------------------------------------------------------------------------
extern "C" void kernel_launch(void* const* d_in, const int* in_sizes, int n_in,
                              void* d_out, int out_size, void* d_ws, size_t ws_size,
                              hipStream_t stream) {
  (void)in_sizes; (void)n_in; (void)out_size; (void)ws_size;
  const float* x          = (const float*)d_in[0];
  const int*   ei         = (const int*)  d_in[1];
  const int*   src        = ei;
  const int*   dst        = ei + NEDGE;
  const float* in_proj_w  = (const float*)d_in[2];
  const float* in_proj_b  = (const float*)d_in[3];
  const float* out_proj_w = (const float*)d_in[4];
  const float* out_proj_b = (const float*)d_in[5];
  const float* ln1_g      = (const float*)d_in[6];
  const float* ln1_b      = (const float*)d_in[7];
  const float* ffn_w1     = (const float*)d_in[8];
  const float* ffn_b1     = (const float*)d_in[9];
  const float* ffn_w2     = (const float*)d_in[10];
  const float* ffn_b2     = (const float*)d_in[11];
  const float* ln2_g      = (const float*)d_in[12];
  const float* ln2_b      = (const float*)d_in[13];
  const float* gcn1_w     = (const float*)d_in[14];
  const float* gcn1_b     = (const float*)d_in[15];
  const float* gcn2_w     = (const float*)d_in[16];
  const float* gcn2_b     = (const float*)d_in[17];
  const float* gcn3_w     = (const float*)d_in[18];
  const float* gcn3_b     = (const float*)d_in[19];
  float* out = (float*)d_out;

  char* wp = (char*)d_ws;
  auto alloc = [&](size_t bytes) -> char* {
    char* p = wp;
    wp += (bytes + 255) & ~(size_t)255;
    return p;
  };
  _Float16* x16     = (_Float16*)alloc((size_t)NODES * DMODEL * 2);
  _Float16* wqkv16  = (_Float16*)alloc((size_t)3 * DMODEL * DMODEL * 2);
  _Float16* wout16  = (_Float16*)alloc((size_t)DMODEL * DMODEL * 2);
  _Float16* wff1_16 = (_Float16*)alloc((size_t)DFFN * DMODEL * 2);
  _Float16* wff2_16 = (_Float16*)alloc((size_t)DMODEL * DFFN * 2);
  _Float16* wg1_16  = (_Float16*)alloc((size_t)HIDDEN * DMODEL * 2);
  _Float16* wg2_16  = (_Float16*)alloc((size_t)HIDDEN * HIDDEN * 2);
  _Float16* wg3_16  = (_Float16*)alloc((size_t)OUTD * HIDDEN * 2);
  _Float16* qkv16   = (_Float16*)alloc((size_t)NODES * 3 * DMODEL * 2);
  _Float16* vt16    = (_Float16*)alloc((size_t)NODES * DMODEL * 2);
  _Float16* ctx16   = (_Float16*)alloc((size_t)NODES * DMODEL * 2);
  float*    att32   = (float*)   alloc((size_t)NODES * DMODEL * 4);
  float*    y32     = (float*)   alloc((size_t)NODES * DMODEL * 4);
  _Float16* y16     = (_Float16*)alloc((size_t)NODES * DMODEL * 2);
  _Float16* ff1_16  = (_Float16*)alloc((size_t)NODES * DFFN * 2);
  float*    ffo32   = (float*)   alloc((size_t)NODES * DMODEL * 4);
  _Float16* z16     = (_Float16*)alloc((size_t)NODES * DMODEL * 2);
  float*    hpre    = (float*)   alloc((size_t)NODES * HIDDEN * 4);
  float*    hsum    = (float*)   alloc((size_t)NODES * HIDDEN * 4);
  _Float16* h16     = (_Float16*)alloc((size_t)NODES * HIDDEN * 2);
  float*    deg     = (float*)   alloc((size_t)NODES * 4);
  float*    dinv    = (float*)   alloc((size_t)NODES * 4);

  auto cvt = [&](const float* in, _Float16* o, int n) {
    wl_cvt_f32_f16<<<(n + 255) / 256, 256, 0, stream>>>(in, o, n);
  };
  auto gemm = [&](const _Float16* A, const _Float16* W, const float* bias,
                  float* Cf, _Float16* Ch, int Nout, int K, int relu) {
    dim3 grid(Nout / 32, NODES / 256);
    wl_gemm_kernel<<<grid, 256, 0, stream>>>(A, W, bias, Cf, Ch, Nout, K, relu);
  };

  // f32 -> f16 conversions
  cvt(x, x16, NODES * DMODEL);
  cvt(in_proj_w,  wqkv16,  3 * DMODEL * DMODEL);
  cvt(out_proj_w, wout16,  DMODEL * DMODEL);
  cvt(ffn_w1,     wff1_16, DFFN * DMODEL);
  cvt(ffn_w2,     wff2_16, DMODEL * DFFN);
  cvt(gcn1_w,     wg1_16,  HIDDEN * DMODEL);
  cvt(gcn2_w,     wg2_16,  HIDDEN * HIDDEN);
  cvt(gcn3_w,     wg3_16,  OUTD * HIDDEN);

  // --- Transformer encoder layer ---
  gemm(x16, wqkv16, in_proj_b, nullptr, qkv16, 3 * DMODEL, DMODEL, 0);
  wl_vtrans<<<(NODES * DMODEL + 255) / 256, 256, 0, stream>>>(qkv16, vt16);
  wl_attn_kernel<<<(NODES / 16 * NHEAD) / 8, 256, 0, stream>>>(qkv16, vt16, ctx16);
  gemm(ctx16, wout16, out_proj_b, att32, nullptr, DMODEL, DMODEL, 0);
  wl_ln_kernel<<<NODES / 8, 256, 0, stream>>>(x, att32, ln1_g, ln1_b, y32, y16);
  gemm(y16, wff1_16, ffn_b1, nullptr, ff1_16, DFFN, DMODEL, 1);
  gemm(ff1_16, wff2_16, ffn_b2, ffo32, nullptr, DMODEL, DFFN, 0);
  wl_ln_kernel<<<NODES / 8, 256, 0, stream>>>(y32, ffo32, ln2_g, ln2_b, nullptr, z16);

  // --- GCN degree normalization (shared by all 3 layers) ---
  wl_fill<<<(NODES + 255) / 256, 256, 0, stream>>>(deg, 1.0f, NODES);  // self loops
  wl_deg_kernel<<<NEDGE / 256, 256, 0, stream>>>(dst, deg);
  wl_dinv_kernel<<<(NODES + 255) / 256, 256, 0, stream>>>(deg, dinv);

  // --- GCN layer 1: 128 -> 256, ReLU ---
  gemm(z16, wg1_16, nullptr, hpre, nullptr, HIDDEN, DMODEL, 0);
  wl_gcn_init<<<(NODES * HIDDEN + 255) / 256, 256, 0, stream>>>(hpre, dinv, gcn1_b, hsum, HIDDEN);
  wl_gcn_scatter<<<(NEDGE * (HIDDEN / 4) + 255) / 256, 256, 0, stream>>>(hpre, dinv, src, dst, hsum, HIDDEN);
  wl_relu_cvt<<<(NODES * HIDDEN + 255) / 256, 256, 0, stream>>>(hsum, h16, NODES * HIDDEN);

  // --- GCN layer 2: 256 -> 256, ReLU ---
  gemm(h16, wg2_16, nullptr, hpre, nullptr, HIDDEN, HIDDEN, 0);
  wl_gcn_init<<<(NODES * HIDDEN + 255) / 256, 256, 0, stream>>>(hpre, dinv, gcn2_b, hsum, HIDDEN);
  wl_gcn_scatter<<<(NEDGE * (HIDDEN / 4) + 255) / 256, 256, 0, stream>>>(hpre, dinv, src, dst, hsum, HIDDEN);
  wl_relu_cvt<<<(NODES * HIDDEN + 255) / 256, 256, 0, stream>>>(hsum, h16, NODES * HIDDEN);

  // --- GCN layer 3: 256 -> 64, no activation; accumulate directly in d_out ---
  gemm(h16, wg3_16, nullptr, hpre, nullptr, OUTD, HIDDEN, 0);
  wl_gcn_init<<<(NODES * OUTD + 255) / 256, 256, 0, stream>>>(hpre, dinv, gcn3_b, out, OUTD);
  wl_gcn_scatter<<<(NEDGE * (OUTD / 4) + 255) / 256, 256, 0, stream>>>(hpre, dinv, src, dst, out, OUTD);
}
